// Mamba_49014166782183
// MI455X (gfx1250) — matrix operand order repaired and checked by reading, not compiled, not run on hardware
//
#include <hip/hip_runtime.h>
#include <hip/hip_bf16.h>

// ---------------------------------------------------------------------------
// Mamba forward (16 layers) for gfx1250 / MI455X.
// GEMMs: v_wmma_f32_16x16x32_f16 on f16 operands, 64x128 block tile,
//        32x32 per wave (4 WMMA / K-step), double-buffered async
//        GLOBAL_LOAD_ASYNC_TO_LDS_B128 staging (one barrier per K-step).
// Scan: register-resident state, one thread per (b, channel).
// ---------------------------------------------------------------------------

#define H_DIM 768
#define I_DIM 1536
#define N_ST  16
#define R_DIM 48
#define K_CV  4
#define NL_   16
#define B_    2
#define L_    1024
#define M_ROWS (B_ * L_)        // 2048  (always a multiple of 64)
#define PROJ_C (R_DIM + 2*N_ST) // 80

typedef __attribute__((ext_vector_type(16))) _Float16 v16h;
typedef __attribute__((ext_vector_type(8)))  _Float16 v8h;
typedef __attribute__((ext_vector_type(4)))  _Float16 v4h;
typedef __attribute__((ext_vector_type(8)))  float    v8f;
typedef __attribute__((ext_vector_type(4)))  int      v4i;

#if __has_builtin(__builtin_amdgcn_global_load_async_to_lds_b128)
#define HAVE_ASYNC_LDS 1
#else
#define HAVE_ASYNC_LDS 0
#endif

// async copy of 16 bytes (8 halves) global -> LDS; falls back to VGPR copy.
// builtin signature: (v4i32 addrspace(1)*, v4i32 addrspace(3)*, imm off, imm cpol)
static __device__ __forceinline__ void stage16B(const _Float16* g, _Float16* l) {
#if HAVE_ASYNC_LDS
    __builtin_amdgcn_global_load_async_to_lds_b128(
        (__attribute__((address_space(1))) v4i*)(_Float16*)g,
        (__attribute__((address_space(3))) v4i*)l, 0, 0);
#else
    *(v8h*)l = *(const v8h*)g;
#endif
}

static __device__ __forceinline__ void stage_wait() {
#if HAVE_ASYNC_LDS
#if __has_builtin(__builtin_amdgcn_s_wait_asynccnt)
    __builtin_amdgcn_s_wait_asynccnt(0);
#else
    asm volatile("s_wait_asynccnt 0x0" ::: "memory");
#endif
#endif
}

static __device__ __forceinline__ float silu_f(float v) {
    return v / (1.0f + __expf(-v));
}

static __device__ __forceinline__ v16h cat16(v8h a, v8h b) {
    v16h r;
    #pragma unroll
    for (int e = 0; e < 8; ++e) { r[e] = a[e]; r[e + 8] = b[e]; }
    return r;
}

// ---------------------------------------------------------------------------
// f32 -> f16 conversion (n4 = element count / 4)
// ---------------------------------------------------------------------------
__global__ void cvt_f16_kernel(const float* __restrict__ src,
                               _Float16* __restrict__ dst, int n4) {
    int idx = blockIdx.x * blockDim.x + threadIdx.x;
    if (idx >= n4) return;
    float4 f = ((const float4*)src)[idx];
    v4h h = { (_Float16)f.x, (_Float16)f.y, (_Float16)f.z, (_Float16)f.w };
    ((v4h*)dst)[idx] = h;
}

// ---------------------------------------------------------------------------
// RMSNorm: one block (256 threads) per row; writes f32 and/or f16.
// ---------------------------------------------------------------------------
__global__ void rmsnorm_kernel(const float* __restrict__ x,
                               const float* __restrict__ w,
                               float* __restrict__ out,
                               _Float16* __restrict__ out16, int Hdim) {
    int row = blockIdx.x;
    const float* xr = x + (size_t)row * Hdim;
    float ss = 0.0f;
    for (int c = threadIdx.x; c < Hdim; c += blockDim.x) {
        float v = xr[c];
        ss += v * v;
    }
    #pragma unroll
    for (int o = 16; o > 0; o >>= 1) ss += __shfl_down(ss, o, 32);
    __shared__ float red[8];
    if ((threadIdx.x & 31) == 0) red[threadIdx.x >> 5] = ss;
    __syncthreads();
    float tot = 0.0f;
    #pragma unroll
    for (int i = 0; i < 8; ++i) tot += red[i];
    float scale = rsqrtf(tot / (float)Hdim + 1e-5f);
    for (int c = threadIdx.x; c < Hdim; c += blockDim.x) {
        float v = xr[c] * scale * w[c];
        if (out)   out[(size_t)row * Hdim + c]   = v;
        if (out16) out16[(size_t)row * Hdim + c] = (_Float16)v;
    }
}

// ---------------------------------------------------------------------------
// WMMA GEMM: C[M,N] = A[M,K] * W[N,K]^T, A/W f16 in global, C f32 (+opt f16).
// Block: 256 threads = 8 waves (2 M x 4 N); block tile 64x128.
// Wave: 32x32 sub-tile = 2x2 accumulators, 4 WMMAs per K-step.
// Double-buffered LDS: async-stage tile k+1 while WMMAs consume tile k;
// one barrier per K-step. Requires M % 64 == 0 (true: M = 2048).
// ---------------------------------------------------------------------------
#define LDS_STRIDE 40  // halves; 80 B row pitch, multiple of 16 B

__global__ void gemm_wmma_kernel(const _Float16* __restrict__ A,
                                 const _Float16* __restrict__ W,
                                 float* __restrict__ C,
                                 _Float16* __restrict__ C16,
                                 int Mr, int Nr, int Kr,
                                 int lda, int ldw, int ldc) {
    __shared__ __align__(16) _Float16 sA[2][64 * LDS_STRIDE];
    __shared__ __align__(16) _Float16 sW[2][128 * LDS_STRIDE];

    const int n0   = blockIdx.x * 128;
    const int m0   = blockIdx.y * 64;
    const int tid  = threadIdx.x;
    const int lane = tid & 31;
    const int wv   = tid >> 5;
    const int wm0  = (wv & 1) * 32;        // wave M offset in block tile
    const int wn0  = (wv >> 1) * 32;       // wave N offset in block tile
    const int half = lane >> 4;
    const int l16  = lane & 15;

    // staging coordinates (A: 8 halves/thread, W: 16 halves/thread)
    const int rA = tid >> 2;               // 0..63
    const int cA = (tid & 3) * 8;          // 0,8,16,24
    const int rW = tid >> 1;               // 0..127
    const int cW = (tid & 1) * 16;         // 0,16

    const bool fullN = (n0 + 128 <= Nr);

    // stage one 64x32 A tile + 128x32 W tile at K offset k0 into buffer b
    auto stage_tile = [&](int b, int k0) {
        const bool fullK = (k0 + 32 <= Kr);
        if (fullN && fullK) {
            // fast path: async b128 copies, no VGPR round-trip
            stage16B(&A[(size_t)(m0 + rA) * lda + k0 + cA],
                     &sA[b][rA * LDS_STRIDE + cA]);
            stage16B(&W[(size_t)(n0 + rW) * ldw + k0 + cW],
                     &sW[b][rW * LDS_STRIDE + cW]);
            stage16B(&W[(size_t)(n0 + rW) * ldw + k0 + cW + 8],
                     &sW[b][rW * LDS_STRIDE + cW + 8]);
        } else {
            // tail path: clamped addresses + branch-free select
            {
                int gm = m0 + rA;                      // always < Mr (M%64==0)
                #pragma unroll
                for (int j = 0; j < 8; ++j) {
                    int gk  = k0 + cA + j;
                    int gkc = gk < Kr ? gk : (Kr - 1);
                    _Float16 v = A[(size_t)gm * lda + gkc];
                    v = (gk < Kr) ? v : (_Float16)0.0f;
                    sA[b][rA * LDS_STRIDE + cA + j] = v;
                }
            }
            {
                int gn  = n0 + rW;
                int gnc = gn < Nr ? gn : (Nr - 1);
                #pragma unroll
                for (int j = 0; j < 16; ++j) {
                    int gk  = k0 + cW + j;
                    int gkc = gk < Kr ? gk : (Kr - 1);
                    _Float16 v = W[(size_t)gnc * ldw + gkc];
                    v = (gn < Nr && gk < Kr) ? v : (_Float16)0.0f;
                    sW[b][rW * LDS_STRIDE + cW + j] = v;
                }
            }
        }
    };

    v8f acc00 = {}, acc01 = {}, acc10 = {}, acc11 = {};

    // prologue: stage tile 0
    stage_tile(0, 0);
    stage_wait();
    __syncthreads();

    int cur = 0;
    for (int k0 = 0; k0 < Kr; k0 += 32) {
        const int nxt = cur ^ 1;

        // kick off async staging of the next tile while we compute this one
        if (k0 + 32 < Kr) stage_tile(nxt, k0 + 32);

        // ---- A fragments (16x32 f16, ISA layout) for the two M-tiles ----
        // lanes 0-15: elems 0..7 -> K 0..7,  8..15 -> K 16..23
        // lanes16-31: elems 0..7 -> K 8..15, 8..15 -> K 24..31
        const int kbA = half * 8;
        const int ra0 = (wm0 + l16) * LDS_STRIDE;
        const int ra1 = (wm0 + 16 + l16) * LDS_STRIDE;
        v16h af0 = cat16(*(const v8h*)&sA[cur][ra0 + kbA],
                         *(const v8h*)&sA[cur][ra0 + 16 + kbA]);
        v16h af1 = cat16(*(const v8h*)&sA[cur][ra1 + kbA],
                         *(const v8h*)&sA[cur][ra1 + 16 + kbA]);

        // ---- B fragments (32x16 f16): same K order as A, N=lane&15 --
        const int kbB = half * 8;
        const int rb0 = (wn0 + l16) * LDS_STRIDE;
        const int rb1 = (wn0 + 16 + l16) * LDS_STRIDE;
        v16h bf0 = cat16(*(const v8h*)&sW[cur][rb0 + kbB],
                         *(const v8h*)&sW[cur][rb0 + 16 + kbB]);
        v16h bf1 = cat16(*(const v8h*)&sW[cur][rb1 + kbB],
                         *(const v8h*)&sW[cur][rb1 + 16 + kbB]);

        acc00 = __builtin_amdgcn_wmma_f32_16x16x32_f16(false, af0, false, bf0,
                                                       (short)0, acc00, false, false);
        acc01 = __builtin_amdgcn_wmma_f32_16x16x32_f16(false, af0, false, bf1,
                                                       (short)0, acc01, false, false);
        acc10 = __builtin_amdgcn_wmma_f32_16x16x32_f16(false, af1, false, bf0,
                                                       (short)0, acc10, false, false);
        acc11 = __builtin_amdgcn_wmma_f32_16x16x32_f16(false, af1, false, bf1,
                                                       (short)0, acc11, false, false);

        // next tile staged (async done) and every wave finished its DS reads
        stage_wait();
        __syncthreads();
        cur = nxt;
    }

    // ---- store: elem r -> M = r + (lane>=16 ? 8 : 0), N = lane&15 ----
    {
        const int mA = m0 + wm0 + half * 8;
        const int mB = m0 + wm0 + 16 + half * 8;
        const int nA = n0 + wn0 + l16;
        const int nB = n0 + wn0 + 16 + l16;
        if (nA < Nr) {
            #pragma unroll
            for (int r = 0; r < 8; ++r) {
                C[(size_t)(mA + r) * ldc + nA] = acc00[r];
                C[(size_t)(mB + r) * ldc + nA] = acc10[r];
            }
            if (C16) {
                #pragma unroll
                for (int r = 0; r < 8; ++r) {
                    C16[(size_t)(mA + r) * ldc + nA] = (_Float16)acc00[r];
                    C16[(size_t)(mB + r) * ldc + nA] = (_Float16)acc10[r];
                }
            }
        }
        if (nB < Nr) {
            #pragma unroll
            for (int r = 0; r < 8; ++r) {
                C[(size_t)(mA + r) * ldc + nB] = acc01[r];
                C[(size_t)(mB + r) * ldc + nB] = acc11[r];
            }
            if (C16) {
                #pragma unroll
                for (int r = 0; r < 8; ++r) {
                    C16[(size_t)(mA + r) * ldc + nB] = (_Float16)acc01[r];
                    C16[(size_t)(mB + r) * ldc + nB] = (_Float16)acc11[r];
                }
            }
        }
    }
}

// ---------------------------------------------------------------------------
// Depthwise causal conv (K=4) + bias + silu; writes f32 (scan) and f16 (GEMM)
// ---------------------------------------------------------------------------
__global__ void conv_silu_kernel(const float* __restrict__ xz,
                                 const float* __restrict__ cw,
                                 const float* __restrict__ cb,
                                 float* __restrict__ u,
                                 _Float16* __restrict__ u16) {
    int idx = blockIdx.x * blockDim.x + threadIdx.x;   // over M*I
    if (idx >= M_ROWS * I_DIM) return;
    int c  = idx % I_DIM;
    int ml = idx / I_DIM;
    int l  = ml % L_;
    int b  = ml / L_;
    float acc = cb[c];
    #pragma unroll
    for (int k = 0; k < K_CV; ++k) {
        int ls  = l - (K_CV - 1) + k;
        int lsc = ls >= 0 ? ls : 0;
        float v = xz[(size_t)(b * L_ + lsc) * (2 * I_DIM) + c];
        v = (ls >= 0) ? v : 0.0f;
        acc += cw[c * K_CV + k] * v;
    }
    float s = silu_f(acc);
    u[idx]   = s;
    u16[idx] = (_Float16)s;
}

// ---------------------------------------------------------------------------
// dt = softplus(dt_raw + bias)   (in-place)
// ---------------------------------------------------------------------------
__global__ void softplus_bias_kernel(float* __restrict__ dt,
                                     const float* __restrict__ dtb) {
    int idx = blockIdx.x * blockDim.x + threadIdx.x;   // over M*I
    if (idx >= M_ROWS * I_DIM) return;
    int c = idx % I_DIM;
    float v = dt[idx] + dtb[c];
    dt[idx] = (v > 20.0f) ? v : __logf(1.0f + __expf(v));
}

// ---------------------------------------------------------------------------
// Selective scan: one thread per (b, i); state h[16] in registers.
// ---------------------------------------------------------------------------
__global__ void scan_kernel(const float* __restrict__ u,
                            const float* __restrict__ dt,
                            const float* __restrict__ proj,
                            const float* __restrict__ A_log,
                            const float* __restrict__ Dp,
                            float* __restrict__ y) {
    int idx = blockIdx.x * blockDim.x + threadIdx.x;   // over B*I
    if (idx >= B_ * I_DIM) return;
    int i = idx % I_DIM;
    int b = idx / I_DIM;
    float Arow[N_ST], h[N_ST];
    #pragma unroll
    for (int n = 0; n < N_ST; ++n) {
        Arow[n] = -__expf(A_log[i * N_ST + n]);
        h[n] = 0.0f;
    }
    float Di = Dp[i];
    for (int t = 0; t < L_; ++t) {
        int row = b * L_ + t;
        float ut  = u[(size_t)row * I_DIM + i];
        float dtt = dt[(size_t)row * I_DIM + i];
        // prefetch next timestep's proj row (global_prefetch_b8)
        if (t + 1 < L_)
            __builtin_prefetch(proj + (size_t)(row + 1) * PROJ_C + R_DIM, 0, 1);
        // proj row is 80 floats; offsets 48 and 64 are 16B-aligned -> b128 loads
        const float4* prB = (const float4*)(proj + (size_t)row * PROJ_C + R_DIM);
        const float4* prC = (const float4*)(proj + (size_t)row * PROJ_C + R_DIM + N_ST);
        float4 bq[4] = { prB[0], prB[1], prB[2], prB[3] };
        float4 cq[4] = { prC[0], prC[1], prC[2], prC[3] };
        const float* Bt = (const float*)bq;
        const float* Ct = (const float*)cq;
        float du = dtt * ut;
        float yv = 0.0f;
        #pragma unroll
        for (int n = 0; n < N_ST; ++n) {
            float dA = __expf(dtt * Arow[n]);
            float hn = dA * h[n] + du * Bt[n];
            h[n] = hn;
            yv += hn * Ct[n];
        }
        y[(size_t)row * I_DIM + i] = yv + ut * Di;
    }
}

// ---------------------------------------------------------------------------
// yz16 = (f16)(y * silu(z)) where z = second half of xz channels
// ---------------------------------------------------------------------------
__global__ void gate_kernel(const float* __restrict__ y,
                            const float* __restrict__ xz,
                            _Float16* __restrict__ yz16) {
    int idx = blockIdx.x * blockDim.x + threadIdx.x;   // over M*I
    if (idx >= M_ROWS * I_DIM) return;
    int c   = idx % I_DIM;
    int row = idx / I_DIM;
    float z = xz[(size_t)row * (2 * I_DIM) + I_DIM + c];
    yz16[idx] = (_Float16)(y[idx] * silu_f(z));
}

// ---------------------------------------------------------------------------
// x += out (residual)
// ---------------------------------------------------------------------------
__global__ void residual_kernel(float* __restrict__ x,
                                const float* __restrict__ o) {
    int idx = blockIdx.x * blockDim.x + threadIdx.x;   // over M*H
    if (idx >= M_ROWS * H_DIM) return;
    x[idx] += o[idx];
}

// ---------------------------------------------------------------------------
static inline int cdiv(int a, int b) { return (a + b - 1) / b; }

extern "C" void kernel_launch(void* const* d_in, const int* in_sizes, int n_in,
                              void* d_out, int out_size, void* d_ws, size_t ws_size,
                              hipStream_t stream) {
    const float* x            = (const float*)d_in[0];
    const float* in_proj_w    = (const float*)d_in[1];
    const float* conv_w       = (const float*)d_in[2];
    const float* conv_b       = (const float*)d_in[3];
    const float* x_proj_w     = (const float*)d_in[4];
    const float* dt_proj_w    = (const float*)d_in[5];
    const float* dt_proj_b    = (const float*)d_in[6];
    const float* A_log        = (const float*)d_in[7];
    const float* Dvec         = (const float*)d_in[8];
    const float* out_proj_w   = (const float*)d_in[9];
    const float* norm_w       = (const float*)d_in[10];
    const float* final_norm_w = (const float*)d_in[11];

    // ---- workspace partition ----
    float* ws = (float*)d_ws;
    size_t off = 0;   // in floats
    float* x_cur = ws + off; off += (size_t)M_ROWS * H_DIM;
    float* xz    = ws + off; off += (size_t)M_ROWS * 2 * I_DIM;
    float* u     = ws + off; off += (size_t)M_ROWS * I_DIM;
    float* proj  = ws + off; off += (size_t)M_ROWS * PROJ_C;
    float* dt    = ws + off; off += (size_t)M_ROWS * I_DIM;
    float* y     = ws + off; off += (size_t)M_ROWS * I_DIM;
    float* outb  = ws + off; off += (size_t)M_ROWS * H_DIM;
    // f16 region (2 halves per float slot)
    _Float16* h16 = (_Float16*)(ws + off);
    size_t hoff = 0;  // in halves
    _Float16* hbuf16 = h16 + hoff; hoff += (size_t)M_ROWS * H_DIM;
    _Float16* u16    = h16 + hoff; hoff += (size_t)M_ROWS * I_DIM;
    _Float16* proj16 = h16 + hoff; hoff += (size_t)M_ROWS * PROJ_C;
    _Float16* yz16   = h16 + hoff; hoff += (size_t)M_ROWS * I_DIM;
    _Float16* w16_in = h16 + hoff; hoff += (size_t)2 * I_DIM * H_DIM;
    _Float16* w16_xp = h16 + hoff; hoff += (size_t)PROJ_C * I_DIM;
    _Float16* w16_dt = h16 + hoff; hoff += (size_t)I_DIM * R_DIM;
    _Float16* w16_ow = h16 + hoff; hoff += (size_t)H_DIM * I_DIM;
    (void)ws_size; (void)in_sizes; (void)n_in; (void)out_size;

    (void)hipMemcpyAsync(x_cur, x, (size_t)M_ROWS * H_DIM * sizeof(float),
                         hipMemcpyDeviceToDevice, stream);

    const int elemI = M_ROWS * I_DIM;
    const int elemH = M_ROWS * H_DIM;

    for (int layer = 0; layer < NL_; ++layer) {
        const float* in_w  = in_proj_w  + (size_t)layer * 2 * I_DIM * H_DIM;
        const float* cw    = conv_w     + (size_t)layer * I_DIM * K_CV;
        const float* cb    = conv_b     + (size_t)layer * I_DIM;
        const float* xp_w  = x_proj_w   + (size_t)layer * PROJ_C * I_DIM;
        const float* dtw   = dt_proj_w  + (size_t)layer * I_DIM * R_DIM;
        const float* dtb   = dt_proj_b  + (size_t)layer * I_DIM;
        const float* alog  = A_log      + (size_t)layer * I_DIM * N_ST;
        const float* dvec  = Dvec       + (size_t)layer * I_DIM;
        const float* ow    = out_proj_w + (size_t)layer * H_DIM * I_DIM;
        const float* nw    = norm_w     + (size_t)layer * H_DIM;

        // ---- convert this layer's weights to f16 ----
        cvt_f16_kernel<<<cdiv(2 * I_DIM * H_DIM / 4, 256), 256, 0, stream>>>(
            in_w, w16_in, 2 * I_DIM * H_DIM / 4);
        cvt_f16_kernel<<<cdiv(PROJ_C * I_DIM / 4, 256), 256, 0, stream>>>(
            xp_w, w16_xp, PROJ_C * I_DIM / 4);
        cvt_f16_kernel<<<cdiv(I_DIM * R_DIM / 4, 256), 256, 0, stream>>>(
            dtw, w16_dt, I_DIM * R_DIM / 4);
        cvt_f16_kernel<<<cdiv(H_DIM * I_DIM / 4, 256), 256, 0, stream>>>(
            ow, w16_ow, H_DIM * I_DIM / 4);

        // h = rmsnorm(x, nw) -> f16
        rmsnorm_kernel<<<M_ROWS, 256, 0, stream>>>(x_cur, nw, nullptr, hbuf16, H_DIM);

        // xz = h @ in_w^T : [2048,3072] = [2048,768] x [3072,768]^T
        gemm_wmma_kernel<<<dim3(cdiv(2 * I_DIM, 128), cdiv(M_ROWS, 64)), 256, 0, stream>>>(
            hbuf16, w16_in, xz, nullptr,
            M_ROWS, 2 * I_DIM, H_DIM, H_DIM, H_DIM, 2 * I_DIM);

        // u = silu(causal_conv(xz[:, :I])) -> f32 + f16
        conv_silu_kernel<<<cdiv(elemI, 256), 256, 0, stream>>>(xz, cw, cb, u, u16);

        // proj = u @ xp_w^T : [2048,80] = [2048,1536] x [80,1536]^T -> f32 + f16
        gemm_wmma_kernel<<<dim3(cdiv(PROJ_C, 128), cdiv(M_ROWS, 64)), 256, 0, stream>>>(
            u16, w16_xp, proj, proj16,
            M_ROWS, PROJ_C, I_DIM, I_DIM, I_DIM, PROJ_C);

        // dt_raw = proj[:, :48] @ dtw^T : [2048,1536] = [2048,48(lda=80)] x [1536,48]^T
        gemm_wmma_kernel<<<dim3(cdiv(I_DIM, 128), cdiv(M_ROWS, 64)), 256, 0, stream>>>(
            proj16, w16_dt, dt, nullptr,
            M_ROWS, I_DIM, R_DIM, PROJ_C, R_DIM, I_DIM);

        // dt = softplus(dt_raw + dtb)
        softplus_bias_kernel<<<cdiv(elemI, 256), 256, 0, stream>>>(dt, dtb);

        // y = selective_scan(u, dt, A, B, C, D)
        scan_kernel<<<cdiv(B_ * I_DIM, 256), 256, 0, stream>>>(
            u, dt, proj, alog, dvec, y);

        // yz16 = y * silu(z)
        gate_kernel<<<cdiv(elemI, 256), 256, 0, stream>>>(y, xz, yz16);

        // out = yz @ ow^T : [2048,768] = [2048,1536] x [768,1536]^T
        gemm_wmma_kernel<<<dim3(cdiv(H_DIM, 128), cdiv(M_ROWS, 64)), 256, 0, stream>>>(
            yz16, w16_ow, outb, nullptr,
            M_ROWS, H_DIM, I_DIM, I_DIM, I_DIM, H_DIM);

        // x += out
        residual_kernel<<<cdiv(elemH, 256), 256, 0, stream>>>(x_cur, outb);
    }

    // final rmsnorm -> d_out (f32)
    rmsnorm_kernel<<<M_ROWS, 256, 0, stream>>>(x_cur, final_norm_w,
                                               (float*)d_out, nullptr, H_DIM);
}
